// GCN_fusion6_91036126806365
// MI455X (gfx1250) — matrix operand order repaired and checked by reading, not compiled, not run on hardware
//
#include <hip/hip_runtime.h>
#include <math.h>

typedef __attribute__((ext_vector_type(16))) _Float16 v16h;
typedef __attribute__((ext_vector_type(8)))  float    v8f;

// ---------------------------------------------------------------------------
// Zero-fill (ws is poisoned 0xAA by the harness; atomics need clean zeros)
// ---------------------------------------------------------------------------
__global__ void zero_kernel(float* __restrict__ p, long long n) {
    long long i = (long long)blockIdx.x * blockDim.x + threadIdx.x;
    long long stride = (long long)gridDim.x * blockDim.x;
    for (; i < n; i += stride) p[i] = 0.0f;
}

// ---------------------------------------------------------------------------
// Pack fp32 B[K,N] into f16 WMMA B-fragments (ISA 32x16 16-bit layout):
//   Bp[((nt*Ktiles + kt)*32 + lane)*16 + j]
//     lane<16 : column n0+lane,    K = k0+(j>>3)*16+(j&7)     (j = 0..15)
//     lane>=16: column n0+lane-16, K = k0+(j>>3)*16+8+(j&7)
// Done once per weight matrix; weights are reused by ~3125 M-tiles, so the
// GEMM inner loop gets a single contiguous 32B load instead of 16 strided ones.
// ---------------------------------------------------------------------------
__global__ void pack_b_kernel(const float* __restrict__ B,
                              _Float16* __restrict__ Bp, int K, int N) {
    int idx = blockIdx.x * blockDim.x + threadIdx.x;
    if (idx >= K * N) return;
    const int Ktiles = K >> 5;
    const int j    = idx & 15;
    const int lane = (idx >> 4) & 31;
    const int t    = idx >> 9;              // nt*Ktiles + kt
    const int kt   = t % Ktiles;
    const int nt   = t / Ktiles;
    const int col  = nt * 16 + (lane & 15);
    const int row  = kt * 32 + (j >> 3) * 16 + (lane >> 4) * 8 + (j & 7);
    Bp[idx] = (_Float16)B[(size_t)row * N + col];
}

// ---------------------------------------------------------------------------
// C[M,N] = A[M,K] * Bp   (A fp32, Bp pre-packed f16 fragments, fp32 accum)
// Requirements: M % 16 == 0, K % 32 == 0, N % 16 == 0 (all hold here).
// One wave per 16-row strip of C (loops over all N/16 tiles); 8 waves per block. EXEC all-ones inside WMMA.
// ---------------------------------------------------------------------------
__global__ __launch_bounds__(256)
void gemm_wmma_f16(const float* __restrict__ A, const v16h* __restrict__ Bp,
                   float* __restrict__ C, int M, int K, int N) {
    const int lane  = threadIdx.x & 31;
    const int wave  = threadIdx.x >> 5;
    const int tileM = blockIdx.x * 8 + wave;
    const int row0  = tileM * 16;
    if (row0 >= M) return;                 // uniform per wave

    const int l15 = lane & 15;
    const int lhi = lane >> 4;             // 0 or 1
    const int Ktiles = K >> 5;

    const float* arow = A + (size_t)(row0 + l15) * K;

    for (int nt = 0; nt * 16 < N; ++nt) {
        v8f acc = {};
        const v16h* bbase = Bp + (size_t)nt * Ktiles * 32 + lane;
        for (int kt = 0; kt < Ktiles; ++kt) {
            // ---- A fragment: 16x32, ISA 16-bit A layout (convert on the fly)
            const float* ap = arow + kt * 32 + lhi * 8;
            __builtin_prefetch(ap + 32, 0, 1);   // global_prefetch_b8 next K-block
            v16h a;
#pragma unroll
            for (int j = 0; j < 8; ++j) a[j]     = (_Float16)ap[j];
#pragma unroll
            for (int j = 0; j < 8; ++j) a[8 + j] = (_Float16)ap[16 + j];

            // ---- B fragment: one aligned 32B load of the pre-packed tile
            v16h b = bbase[(size_t)kt * 32];

            acc = __builtin_amdgcn_wmma_f32_16x16x32_f16(
                      false, a, false, b, (short)0, acc, false, false);
        }
        // ---- C store: VGPR r -> row r (lanes 0-15) / row 8+r (lanes 16-31)
        float* crow = C + (size_t)(row0 + (lhi << 3)) * N + nt * 16 + l15;
#pragma unroll
        for (int r = 0; r < 8; ++r) crow[(size_t)r * N] = acc[r];
    }
}

// ---------------------------------------------------------------------------
// Edge scatter: agg[dst[e], d0..d0+3] += support[src[e], d0..d0+3] * w[e]
// float4 gather + 4x global_atomic_add_f32 per thread.
// ---------------------------------------------------------------------------
__global__ void scatter_kernel(const float* __restrict__ sup,
                               const int* __restrict__ src,
                               const int* __restrict__ dst,
                               const float* __restrict__ ew,
                               float* __restrict__ agg, int E, int D) {
    int e = blockIdx.x * blockDim.x + threadIdx.x;
    if (e >= E) return;
    const int d0 = blockIdx.y * 4;
    const int s  = src[e];
    const int dd = dst[e];
    const float w = ew[e];
    const float4 v = *(const float4*)(sup + (size_t)s * D + d0);
    float* ap = agg + (size_t)dd * D + d0;
    atomicAdd(ap + 0, v.x * w);
    atomicAdd(ap + 1, v.y * w);
    atomicAdd(ap + 2, v.z * w);
    atomicAdd(ap + 3, v.w * w);
}

// ---------------------------------------------------------------------------
// In-place h = relu(agg + b)   (one block per node row, blockDim.x == D)
// ---------------------------------------------------------------------------
__global__ void bias_relu_kernel(float* __restrict__ agg,
                                 const float* __restrict__ b, int D) {
    const size_t i = (size_t)blockIdx.x * D + threadIdx.x;
    agg[i] = fmaxf(agg[i] + b[threadIdx.x], 0.0f);
}

// ---------------------------------------------------------------------------
// colsum[c] += sum over rows of relu(agg2[r,c] + b2[c])    (D == 64)
// ---------------------------------------------------------------------------
__global__ __launch_bounds__(256)
void colsum_kernel(const float* __restrict__ agg, const float* __restrict__ b,
                   float* __restrict__ colsum, int Nrows) {
    const int c    = threadIdx.x & 63;
    const int rsub = threadIdx.x >> 6;          // 0..3
    const int row0 = blockIdx.x * 256;
    const float bc = b[c];
    float acc = 0.0f;
    const int rend = min(row0 + 256, Nrows);
    for (int r = row0 + rsub; r < rend; r += 4)
        acc += fmaxf(agg[(size_t)r * 64 + c] + bc, 0.0f);
    __shared__ float sh[256];
    sh[threadIdx.x] = acc;
    __syncthreads();
    if (rsub == 0)
        atomicAdd(&colsum[c], sh[c] + sh[64 + c] + sh[128 + c] + sh[192 + c]);
}

// ---------------------------------------------------------------------------
// Final: pooled = selu(colsum/N); x_ext = sub_fea @ fc_w^T + fc_b;
//        out = log_softmax(pooled + 0.5 * x_ext)        (64 outputs, 1 block)
// ---------------------------------------------------------------------------
__global__ void final_kernel(const float* __restrict__ colsum,
                             const float* __restrict__ sub_fea,
                             const float* __restrict__ fc_w,
                             const float* __restrict__ fc_b,
                             float* __restrict__ out, float invN) {
    const int t = threadIdx.x;          // 64 threads
    const float SELU_SCALE = 1.0507009873554805f;
    const float SELU_ALPHA = 1.6732632423543772f;

    float m = colsum[t] * invN;
    float pooled = (m > 0.0f) ? SELU_SCALE * m
                              : SELU_SCALE * SELU_ALPHA * expm1f(m);
    float xext = 0.0f;
    const float* wrow = fc_w + (size_t)t * 128;
#pragma unroll 8
    for (int j = 0; j < 128; ++j) xext += sub_fea[j] * wrow[j];
    xext += fc_b[t];

    float o = pooled + 0.5f * xext;

    __shared__ float sh[64];
    sh[t] = o;
    __syncthreads();
    float mx = -INFINITY;
    for (int i = 0; i < 64; ++i) mx = fmaxf(mx, sh[i]);
    float se = 0.0f;
    for (int i = 0; i < 64; ++i) se += expf(sh[i] - mx);
    out[t] = o - mx - logf(se);
}

// ---------------------------------------------------------------------------
extern "C" void kernel_launch(void* const* d_in, const int* in_sizes, int n_in,
                              void* d_out, int out_size, void* d_ws, size_t ws_size,
                              hipStream_t stream) {
    const float* x    = (const float*)d_in[0];
    const int*   ei   = (const int*)  d_in[1];
    const float* ew   = (const float*)d_in[2];
    const float* sub  = (const float*)d_in[3];
    const float* W1   = (const float*)d_in[4];
    const float* b1   = (const float*)d_in[5];
    const float* W2   = (const float*)d_in[6];
    const float* b2   = (const float*)d_in[7];
    const float* fcw  = (const float*)d_in[8];
    const float* fcb  = (const float*)d_in[9];
    float* out = (float*)d_out;

    const int D1 = in_sizes[5];              // 96
    const int D2 = in_sizes[7];              // 64
    const int K0 = in_sizes[4] / D1;         // 256
    const int Nn = in_sizes[0] / K0;         // 50000 (multiple of 16)
    const int E  = in_sizes[2];              // 800000

    // ws layout (all chunks 256B-aligned; v16h fragment loads need 32B align)
    char* ws = (char*)d_ws;
    float* support1 = (float*)ws;  ws += (size_t)Nn * D1 * sizeof(float);
    float* agg1     = (float*)ws;  ws += (size_t)Nn * D1 * sizeof(float);
    float* support2 = (float*)ws;  ws += (size_t)Nn * D2 * sizeof(float);
    float* agg2     = (float*)ws;  ws += (size_t)Nn * D2 * sizeof(float);
    float* colsum   = (float*)ws;  ws += 256;
    _Float16* W1p   = (_Float16*)ws;  ws += (size_t)K0 * D1 * sizeof(_Float16);
    _Float16* W2p   = (_Float16*)ws;

    const int* src = ei;
    const int* dst = ei + E;

    // 0) clear accumulators + pack weights into WMMA B-fragment layout
    zero_kernel<<<512, 256, 0, stream>>>(agg1, (long long)Nn * D1);
    zero_kernel<<<512, 256, 0, stream>>>(agg2, (long long)Nn * D2);
    zero_kernel<<<1, 64, 0, stream>>>(colsum, (long long)D2);
    pack_b_kernel<<<(K0 * D1 + 255) / 256, 256, 0, stream>>>(W1, W1p, K0, D1);
    pack_b_kernel<<<(D1 * D2 + 255) / 256, 256, 0, stream>>>(W2, W2p, D1, D2);

    const int tilesM = (Nn + 15) / 16;
    const int gemmBlocks = (tilesM + 7) / 8;

    // 1) support1 = x @ W1
    gemm_wmma_f16<<<gemmBlocks, 256, 0, stream>>>(x, (const v16h*)W1p,
                                                  support1, Nn, K0, D1);

    // 2) agg1 = scatter-add of edge messages (layer 1)
    {
        dim3 g((E + 255) / 256, D1 / 4);
        scatter_kernel<<<g, 256, 0, stream>>>(support1, src, dst, ew, agg1, E, D1);
    }

    // 3) h1 = relu(agg1 + b1)   (in place)
    bias_relu_kernel<<<Nn, D1, 0, stream>>>(agg1, b1, D1);

    // 4) support2 = h1 @ W2
    gemm_wmma_f16<<<gemmBlocks, 256, 0, stream>>>(agg1, (const v16h*)W2p,
                                                  support2, Nn, D1, D2);

    // 5) agg2 = scatter-add (layer 2)
    {
        dim3 g((E + 255) / 256, D2 / 4);
        scatter_kernel<<<g, 256, 0, stream>>>(support2, src, dst, ew, agg2, E, D2);
    }

    // 6) column sums of relu(agg2 + b2)
    colsum_kernel<<<(Nn + 255) / 256, 256, 0, stream>>>(agg2, b2, colsum, Nn);

    // 7) selu-mean + FC + log_softmax -> d_out (64 values)
    final_kernel<<<1, 64, 0, stream>>>(colsum, sub, fcw, fcb, out, 1.0f / (float)Nn);
}
